// MyModel_87522843560830
// MI455X (gfx1250) — compile-verified, matrix operand order checked by reading, not run on hardware
//
#include <hip/hip_runtime.h>
#include <cmath>

// ---------------------------------------------------------------------------
// Types for CDNA5 WMMA (gfx1250, wave32)
// ---------------------------------------------------------------------------
typedef _Float16 f16;
typedef _Float16 v16h __attribute__((ext_vector_type(16)));
typedef _Float16 v8h  __attribute__((ext_vector_type(8)));
typedef float    v8f  __attribute__((ext_vector_type(8)));

union Frag16 { v16h v; v8h half8[2]; f16 h[16]; };

// LDS byte offset of a __shared__ object (addrspace(3) ptrtoint)
__device__ __forceinline__ unsigned lds_off(const void* p) {
  return (unsigned)(unsigned long long)(const __attribute__((address_space(3))) char*)p;
}

// ---------------------------------------------------------------------------
// f32 -> f16 conversion (weights + input image); zero fill for the zero page
// ---------------------------------------------------------------------------
__global__ void cvt_f32_f16(const float* __restrict__ src, f16* __restrict__ dst, int n) {
  int i = blockIdx.x * blockDim.x + threadIdx.x;
  if (i < n) dst[i] = (f16)src[i];
}

__global__ void zero_u32(unsigned* __restrict__ p, int n) {
  int i = blockIdx.x * blockDim.x + threadIdx.x;
  if (i < n) p[i] = 0u;
}

// ---------------------------------------------------------------------------
// Implicit-GEMM conv, SAME padding, NHWC, f16 in / f32 acc (WMMA) / f16|f32 out
// Block: 256 threads = 8 waves in a 4(M) x 2(N) grid.
// Block tile: 128 output pixels x 64 output channels, K-chunks of 64.
// Per wave:  32x32 tile = 2x2 register tile -> 8 WMMAs per chunk.
// Double-buffered LDS, A tile staged with GLOBAL_LOAD_ASYNC_TO_LDS_B128
// (ASYNCcnt path; padding slots read a zero page so EXEC stays full).
// ---------------------------------------------------------------------------
__global__ __launch_bounds__(256)
void conv_wmma_f16(const f16* __restrict__ in, const f16* __restrict__ wts,
                   const float* __restrict__ bias, const f16* __restrict__ zp,
                   void* __restrict__ outp,
                   int outF32, int relu,
                   int N, int H, int W, int Cin,
                   int Ho, int Wo, int Cout,
                   int K, int stride, int padLo) {
  // row strides are multiples of 16B so vector LDS ops stay aligned
  __shared__ __align__(16) f16 As[2][128][72];   // [buf][pixel][k]  144B rows
  __shared__ __align__(16) f16 BsT[2][64][72];   // [buf][cout][k]   144B rows

  const int tid    = threadIdx.x;
  const int lane   = tid & 31;
  const int wid    = tid >> 5;
  const int waveM  = wid >> 1;       // 0..3
  const int waveN  = wid & 1;        // 0..1
  const int laneLo = lane & 15;
  const int laneHi = lane >> 4;

  const int M     = N * Ho * Wo;
  const int R     = K * K * Cin;
  const int hw    = Ho * Wo;
  const int mBase = blockIdx.x * 128;
  const int nBase = blockIdx.y * 64;
  const bool fast = (Cin & 63) == 0;   // uniform; true for every layer but c1_1

  v8f acc00 = {}, acc01 = {}, acc10 = {}, acc11 = {};

  // ---- staging helpers (block-uniform control flow) ----
  auto stageA = [&](int buf, int rBase) {
    if (fast) {
      const int tap = rBase / Cin;            // chunk lies inside one (ky,kx) tap
      const int ciB = rBase - tap * Cin;      // multiple of 64
      const int kx  = tap % K, ky = tap / K;
      #pragma unroll
      for (int jj = 0; jj < 4; ++jj) {
        int v   = tid + 256 * jj;             // 0..1023 vector slots (16B each)
        int m   = v >> 3;                     // 0..127
        int kk0 = (v & 7) * 8;                // 0..56
        int p   = mBase + m;
        const f16* src = zp;                  // zero page for pad / OOB slots
        if (p < M) {
          int n  = p / hw, rm = p - n * hw;
          int oy = rm / Wo, ox = rm - oy * Wo;
          int iy = oy * stride + ky - padLo;
          int ix = ox * stride + kx - padLo;
          if ((unsigned)iy < (unsigned)H && (unsigned)ix < (unsigned)W)
            src = in + (size_t)((n * H + iy) * W + ix) * Cin + ciB + kk0;
        }
        unsigned dst = lds_off(&As[buf][m][kk0]);
        // CDNA5 async copy: 16B/lane global -> LDS, tracked by ASYNCcnt
        asm volatile("global_load_async_to_lds_b128 %0, %1, off"
                     :: "v"(dst), "v"(src) : "memory");
      }
    } else {                                  // first layer only (Cin==3, R==27)
      for (int j = 0; j < 32; ++j) {
        int e  = tid + 256 * j;               // 0..8191
        int m  = e >> 6;
        int kk = e & 63;
        int r  = rBase + kk;
        int p  = mBase + m;
        f16 v  = (f16)0.f;
        if (r < R && p < M) {
          int ci = r % Cin, t = r / Cin;
          int kx = t % K,  ky = t / K;
          int n  = p / hw, rm = p - n * hw;
          int oy = rm / Wo, ox = rm - oy * Wo;
          int iy = oy * stride + ky - padLo;
          int ix = ox * stride + kx - padLo;
          if ((unsigned)iy < (unsigned)H && (unsigned)ix < (unsigned)W)
            v = in[(size_t)((n * H + iy) * W + ix) * Cin + ci];
        }
        As[buf][m][kk] = v;
      }
    }
  };

  auto stageB = [&](int buf, int rBase) {
    #pragma unroll
    for (int jj = 0; jj < 2; ++jj) {
      int v   = tid + 256 * jj;               // 0..511 vector slots
      int kk  = v >> 3;                       // 0..63
      int cn0 = (v & 7) * 8;                  // 0..56
      int r   = rBase + kk;
      v8h wv  = {};
      if (r < R) wv = *(const v8h*)(wts + (size_t)r * Cout + nBase + cn0);
      #pragma unroll
      for (int q = 0; q < 8; ++q) BsT[buf][cn0 + q][kk] = wv[q];
      if (rBase + 128 + kk < R)               // speculative prefetch, 2 chunks ahead
        __builtin_prefetch(wts + (size_t)(rBase + 128 + kk) * Cout + nBase + cn0, 0, 1);
    }
  };

  auto compute = [&](int buf) {
    #pragma unroll
    for (int s = 0; s < 2; ++s) {             // two 32-wide sub-chunks
      const int kOff = s * 32;
      Frag16 a0, a1, b0, b1;
      const f16* p0 = &As[buf][waveM * 32 + laneLo][kOff + laneHi * 8];
      a0.half8[0] = *(const v8h*)p0;          a0.half8[1] = *(const v8h*)(p0 + 16);
      const f16* p1 = &As[buf][waveM * 32 + 16 + laneLo][kOff + laneHi * 8];
      a1.half8[0] = *(const v8h*)p1;          a1.half8[1] = *(const v8h*)(p1 + 16);
      const f16* q0 = &BsT[buf][waveN * 32 + laneLo][kOff + laneHi * 8];
      b0.half8[0] = *(const v8h*)q0;          b0.half8[1] = *(const v8h*)(q0 + 16);
      const f16* q1 = &BsT[buf][waveN * 32 + 16 + laneLo][kOff + laneHi * 8];
      b1.half8[0] = *(const v8h*)q1;          b1.half8[1] = *(const v8h*)(q1 + 16);
      acc00 = __builtin_amdgcn_wmma_f32_16x16x32_f16(false, a0.v, false, b0.v, (short)0, acc00, false, false);
      acc01 = __builtin_amdgcn_wmma_f32_16x16x32_f16(false, a0.v, false, b1.v, (short)0, acc01, false, false);
      acc10 = __builtin_amdgcn_wmma_f32_16x16x32_f16(false, a1.v, false, b0.v, (short)0, acc10, false, false);
      acc11 = __builtin_amdgcn_wmma_f32_16x16x32_f16(false, a1.v, false, b1.v, (short)0, acc11, false, false);
    }
  };

  // async LDS writes are NOT covered by the barrier's DS wait: drain ASYNCcnt first
  auto wait_async_and_barrier = [&]() {
    asm volatile("s_wait_asynccnt 0x0" ::: "memory");
    __syncthreads();
  };

  // ---- double-buffered main loop: one barrier per 64-wide chunk ----
  stageA(0, 0);
  stageB(0, 0);
  wait_async_and_barrier();
  int buf = 0;
  for (int rBase = 0; rBase < R; rBase += 64) {
    int nxt = rBase + 64;
    if (nxt < R) {                            // stage next chunk into other buffer
      stageA(buf ^ 1, nxt);
      stageB(buf ^ 1, nxt);
    }
    compute(buf);
    wait_async_and_barrier();                 // staging done + LDS reads done
    buf ^= 1;
  }

  // ---- epilogue: D layout row = v + 8*laneHi, col = lane%16 ----
  const int colBase = nBase + waveN * 32 + laneLo;
  const int rowBase = mBase + waveM * 32 + laneHi * 8;
  const float bv0 = (colBase      < Cout) ? bias[colBase]      : 0.f;
  const float bv1 = (colBase + 16 < Cout) ? bias[colBase + 16] : 0.f;

  auto store_tile = [&](const v8f& a, int rb, int col, float bv) {
    if (col >= Cout) return;
    #pragma unroll
    for (int v = 0; v < 8; ++v) {
      int p = rb + v;
      if (p < M) {
        float val = a[v] + bv;
        if (relu) val = fmaxf(val, 0.f);
        size_t o = (size_t)p * Cout + col;
        if (outF32) ((float*)outp)[o] = val;
        else        ((f16*)outp)[o]  = (f16)val;
      }
    }
  };
  store_tile(acc00, rowBase,      colBase,      bv0);
  store_tile(acc01, rowBase,      colBase + 16, bv1);
  store_tile(acc10, rowBase + 16, colBase,      bv0);
  store_tile(acc11, rowBase + 16, colBase + 16, bv1);
}

// ---------------------------------------------------------------------------
// 2x2 maxpool, stride 2, VALID
// ---------------------------------------------------------------------------
__global__ void maxpool2x2(const f16* __restrict__ in, f16* __restrict__ out,
                           int N, int H, int W, int C, int Ho, int Wo) {
  long long i = (long long)blockIdx.x * blockDim.x + threadIdx.x;
  long long total = (long long)N * Ho * Wo * C;
  if (i >= total) return;
  int c  = (int)(i % C);  long long t = i / C;
  int ox = (int)(t % Wo); t /= Wo;
  int oy = (int)(t % Ho); t /= Ho;
  int n  = (int)t;
  const f16* base = in + ((size_t)(n * H + oy * 2) * W + ox * 2) * C + c;
  float a = (float)base[0];
  float b = (float)base[C];
  float d = (float)base[(size_t)W * C];
  float e = (float)base[(size_t)W * C + C];
  out[i] = (f16)fmaxf(fmaxf(a, b), fmaxf(d, e));
}

// ---------------------------------------------------------------------------
// Keys cubic (a = -0.75), clip taps, zero outside-source mask (cv2 semantics)
// ---------------------------------------------------------------------------
__device__ __forceinline__ void cubw(float f, float w[4]) {
  const float a = -0.75f;
  float x0 = f + 1.f;
  w[0] = ((a * x0 - 5.f * a) * x0 + 8.f * a) * x0 - 4.f * a;
  w[1] = ((a + 2.f) * f - (a + 3.f)) * f * f + 1.f;
  float g = 1.f - f;
  w[2] = ((a + 2.f) * g - (a + 3.f)) * g * g + 1.f;
  float x3 = 2.f - f;
  w[3] = ((a * x3 - 5.f * a) * x3 + 8.f * a) * x3 - 4.f * a;
}

__device__ __forceinline__ float cubic_sample(const f16* __restrict__ img,
                                              int H, int W, int C, int c,
                                              float xs, float ys) {
  float x0f = floorf(xs), y0f = floorf(ys);
  float fx = xs - x0f, fy = ys - y0f;
  int x0 = (int)x0f, y0 = (int)y0f;
  float wx[4], wy[4];
  cubw(fx, wx); cubw(fy, wy);
  float acc = 0.f;
  #pragma unroll
  for (int i = 0; i < 4; ++i) {
    int yi = min(max(y0 + i - 1, 0), H - 1);
    #pragma unroll
    for (int j = 0; j < 4; ++j) {
      int xj = min(max(x0 + j - 1, 0), W - 1);
      acc += wy[i] * wx[j] * (float)img[((size_t)yi * W + xj) * C + c];
    }
  }
  bool inside = (xs >= 0.f) && (xs <= (float)(W - 1)) &&
                (ys >= 0.f) && (ys <= (float)(H - 1));
  return inside ? acc : 0.f;
}

// warpPolar(LOG): out pixel (y=angle bin, x=rho bin)
__global__ void warp_logpolar(const f16* __restrict__ in, f16* __restrict__ lp,
                              int N, int C, int S /*src H=W*/, int D /*dst*/,
                              float Klog, float Kang) {
  long long i = (long long)blockIdx.x * blockDim.x + threadIdx.x;
  long long total = (long long)N * D * D * C;
  if (i >= total) return;
  int c = (int)(i % C);  long long t = i / C;
  int x = (int)(t % D);  t /= D;
  int y = (int)(t % D);  t /= D;
  int n = (int)t;
  float rho = expf((float)x / Klog);
  float ang = (float)y / Kang;
  float sx = 0.5f * (float)S + rho * cosf(ang);
  float sy = 0.5f * (float)S + rho * sinf(ang);
  lp[i] = (f16)cubic_sample(in + (size_t)n * S * S * C, S, S, C, c, sx, sy);
}

// cv2.resize-style pixel-center-aligned cubic resize D x D -> S x S
__global__ void resize_cubic(const f16* __restrict__ lp, f16* __restrict__ out,
                             int N, int C, int D, int S) {
  long long i = (long long)blockIdx.x * blockDim.x + threadIdx.x;
  long long total = (long long)N * S * S * C;
  if (i >= total) return;
  int c  = (int)(i % C);  long long t = i / C;
  int ox = (int)(t % S);  t /= S;
  int oy = (int)(t % S);  t /= S;
  int n  = (int)t;
  float sc = (float)D / (float)S;
  float sx = ((float)ox + 0.5f) * sc - 0.5f;
  float sy = ((float)oy + 0.5f) * sc - 0.5f;
  out[i] = (f16)cubic_sample(lp + (size_t)n * D * D * C, D, D, C, c, sx, sy);
}

// ---------------------------------------------------------------------------
// Host launcher
// ---------------------------------------------------------------------------
extern "C" void kernel_launch(void* const* d_in, const int* in_sizes, int n_in,
                              void* d_out, int out_size, void* d_ws, size_t ws_size,
                              hipStream_t stream) {
  (void)in_sizes; (void)n_in; (void)out_size; (void)ws_size;

  struct LayerDef { int k, cin, cout; };
  static const LayerDef Ls[23] = {
    {3,3,64},{3,64,64},
    {3,64,128},{3,128,128},
    {3,128,256},{3,256,256},{3,256,256},
    {3,256,512},{3,512,512},{3,512,512},
    {3,512,512},{3,512,512},{3,512,512},
    {3,512,1024},{1,1024,1024},
    {1,1024,256},{3,256,256},
    {1,256,128},{3,128,256},
    {1,256,128},{3,128,256},
    {1,256,128},{3,128,256},
  };

  size_t woff[23]; size_t wtot = 0;
  for (int i = 0; i < 23; ++i) {
    woff[i] = wtot;
    wtot += (size_t)Ls[i].k * Ls[i].k * Ls[i].cin * Ls[i].cout;
  }

  const size_t SZ_ACT = (size_t)4 * 300 * 300 * 64 * sizeof(f16);  // 46.08 MB
  f16* bufA = (f16*)((char*)d_ws);
  f16* bufB = (f16*)((char*)d_ws + SZ_ACT);
  f16* wh   = (f16*)((char*)d_ws + 2 * SZ_ACT);
  size_t whBytes = (wtot * sizeof(f16) + 255) & ~(size_t)255;
  f16* xh   = (f16*)((char*)d_ws + 2 * SZ_ACT + whBytes);
  size_t xhBytes = ((size_t)4 * 300 * 300 * 3 * sizeof(f16) + 255) & ~(size_t)255;
  f16* zpage = (f16*)((char*)d_ws + 2 * SZ_ACT + whBytes + xhBytes);  // 256B zero page

  zero_u32<<<1, 64, 0, stream>>>((unsigned*)zpage, 64);

  for (int i = 0; i < 23; ++i) {
    int n = Ls[i].k * Ls[i].k * Ls[i].cin * Ls[i].cout;
    cvt_f32_f16<<<(n + 255) / 256, 256, 0, stream>>>(
        (const float*)d_in[1 + 2 * i], wh + woff[i], n);
  }
  {
    int n = 4 * 300 * 300 * 3;
    cvt_f32_f16<<<(n + 255) / 256, 256, 0, stream>>>((const float*)d_in[0], xh, n);
  }

  auto conv = [&](const f16* in, int H, int li, void* out, bool outF32,
                  bool relu, int stride) -> int {
    const LayerDef& L = Ls[li];
    int Ho = (H + stride - 1) / stride;                 // SAME
    int padTot = (Ho - 1) * stride + L.k - H;
    if (padTot < 0) padTot = 0;
    int padLo = padTot / 2;                             // XLA SAME: lo = floor
    int M = 4 * Ho * Ho;
    dim3 grid((M + 127) / 128, L.cout / 64);
    conv_wmma_f16<<<grid, 256, 0, stream>>>(
        in, wh + woff[li], (const float*)d_in[2 + 2 * li], zpage, out,
        outF32 ? 1 : 0, relu ? 1 : 0,
        4, H, H, L.cin, Ho, Ho, L.cout, L.k, stride, padLo);
    return Ho;
  };
  auto pool = [&](const f16* in, int H, int C, f16* out) -> int {
    int Ho = H / 2;
    long long total = 4LL * Ho * Ho * C;
    maxpool2x2<<<(unsigned)((total + 255) / 256), 256, 0, stream>>>(
        in, out, 4, H, H, C, Ho, Ho);
    return Ho;
  };

  // ---- VGG trunk (ReLU) ----
  conv(xh,   300, 0, bufA, false, true, 1);
  conv(bufA, 300, 1, bufB, false, true, 1);
  int H = pool(bufB, 300, 64, bufA);                    // 150
  conv(bufA, H, 2, bufB, false, true, 1);
  conv(bufB, H, 3, bufA, false, true, 1);
  H = pool(bufA, H, 128, bufB);                         // 75
  conv(bufB, H, 4, bufA, false, true, 1);
  conv(bufA, H, 5, bufB, false, true, 1);
  conv(bufB, H, 6, bufA, false, true, 1);
  H = pool(bufA, H, 256, bufB);                         // 37
  conv(bufB, H, 7, bufA, false, true, 1);
  conv(bufA, H, 8, bufB, false, true, 1);
  conv(bufB, H, 9, bufA, false, true, 1);
  H = pool(bufA, H, 512, bufB);                         // 18
  conv(bufB, H, 10, bufA, false, true, 1);
  conv(bufA, H, 11, bufB, false, true, 1);
  conv(bufB, H, 12, bufA, false, true, 1);              // c5_3 -> bufA (18x18x512)

  // ---- log-polar warp (cv2.warpPolar LOG to 100x100, cubic resize back) ----
  {
    const double S = 18.0;
    const int radius = (int)(S * 0.9);                  // 16
    const double Mlp = S / std::log((double)radius);
    const float Klog = (float)(100.0 / std::log(Mlp));
    const float Kang = (float)(100.0 / (2.0 * 3.14159265358979323846));
    long long t1 = 4LL * 100 * 100 * 512;
    warp_logpolar<<<(unsigned)((t1 + 255) / 256), 256, 0, stream>>>(
        bufA, bufB, 4, 512, 18, 100, Klog, Kang);
    long long t2 = 4LL * 18 * 18 * 512;
    resize_cubic<<<(unsigned)((t2 + 255) / 256), 256, 0, stream>>>(
        bufB, bufA, 4, 512, 100, 18);
  }

  // ---- SSD head (no ReLU) ----
  conv(bufA, 18, 13, bufB, false, false, 1);            // fc6   18x18x1024
  conv(bufB, 18, 14, bufA, false, false, 1);            // fc7   18x18x1024
  conv(bufA, 18, 15, bufB, false, false, 1);            // c8_1  18x18x256
  H = conv(bufB, 18, 16, bufA, false, false, 2);        // c8_2  -> 9x9x256
  conv(bufA, H, 17, bufB, false, false, 1);             // c9_1  9x9x128
  H = conv(bufB, H, 18, bufA, false, false, 2);         // c9_2  -> 5x5x256
  conv(bufA, H, 19, bufB, false, false, 1);             // c10_1 5x5x128
  conv(bufB, H, 20, bufA, false, false, 1);             // c10_2 5x5x256
  conv(bufA, H, 21, bufB, false, false, 1);             // c11_1 5x5x128
  conv(bufB, H, 22, d_out, true, false, 1);             // c11_2 -> f32 (4,5,5,256)
}
